// ChannelCrossAttention_56229711839609
// MI455X (gfx1250) — matrix operand order repaired and checked by reading, not compiled, not run on hardware
//
#include <hip/hip_runtime.h>
#include <hip/hip_bf16.h>

// ---------------------------------------------------------------------------
// XCA cross-attention for MI455X (gfx1250), bf16 WMMA with f32 accumulation.
// Workspace layout (requires ~201 MB):
//   [  0,  64)MB  qT   : bf16 [s][b][512][4096]   (q channel-major)  -- aliased as out2 later
//   [ 64, 128)MB  kT   : bf16 [s][b][512][4096]   (k channel-major)
//   [128, 192)MB  vtk  : bf16 [s][b][4096][512]   (v token-major)
//   [192, 196)MB  WqkvT: bf16 [s][1536][512]
//   [196, 197)MB  WpT  : bf16 [s][512][512]
//   [198, 199)MB  attn : bf16 [s][b][h][64][64]
//   [200MB +   ]  invq, invk : f32 [s][b][512]
// ---------------------------------------------------------------------------

typedef __bf16 bf16;
typedef __attribute__((ext_vector_type(16))) __bf16 v16bf;
typedef __attribute__((ext_vector_type(8)))  __bf16 v8bf;
typedef __attribute__((ext_vector_type(8)))  float  v8f;
typedef __attribute__((ext_vector_type(4)))  float  v4f;

#define DEVINL __device__ __forceinline__

DEVINL v8f zero8() {
  v8f z;
  #pragma unroll
  for (int i = 0; i < 8; ++i) z[i] = 0.f;
  return z;
}

// A fragment (16x32 bf16, M x K), row-major with leading dim lda (elements).
// Per ISA: lane L holds row m = L%16; K chunks [8h..8h+7] and [16+8h..16+8h+7], h=L/16.
DEVINL v16bf frag_a(const bf16* A, int lda, int lane) {
  const int h = lane >> 4, m = lane & 15;
  const bf16* p = A + (size_t)m * lda + 8 * h;
  union { v16bf v; v8bf h2[2]; } u;
  u.h2[0] = *(const v8bf*)p;
  u.h2[1] = *(const v8bf*)(p + 16);
  return u.v;
}

// B fragment (32x16 bf16, K x N) sourced from Bt[N][K] (K contiguous per column).
// Per ISA: lane L holds column n = L%16; K chunks [8h..8h+7] and [16+8h..16+8h+7], h=L/16.
DEVINL v16bf frag_b(const bf16* Bt, int ldb, int lane) {
  const int h = lane >> 4, n = lane & 15;
  const bf16* p = Bt + (size_t)n * ldb + 8 * h;
  union { v16bf v; v8bf h2[2]; } u;
  u.h2[0] = *(const v8bf*)p;
  u.h2[1] = *(const v8bf*)(p + 16);
  return u.v;
}

DEVINL v8f wmma_bf16(v16bf a, v16bf b, v8f c) {
  return __builtin_amdgcn_wmma_f32_16x16x32_bf16(false, a, false, b, (short)0, c,
                                                 false, false);
}

// ---------------------------------------------------------------------------
// Kernel 0: transpose + convert weights to bf16.  WqkvT[s][co][ci], WpT[s][co][ci]
// ---------------------------------------------------------------------------
__global__ void __launch_bounds__(256) k_prep(const float* __restrict__ Wq0,
                                              const float* __restrict__ Wq1,
                                              const float* __restrict__ Wp0,
                                              const float* __restrict__ Wp1,
                                              bf16* __restrict__ WqkvT,
                                              bf16* __restrict__ WpT) {
  const int idx = blockIdx.x * 256 + threadIdx.x;
  const int NQ = 2 * 1536 * 512;
  if (idx < NQ) {
    const int s = idx / (1536 * 512), r = idx % (1536 * 512);
    const int co = r / 512, ci = r % 512;
    const float* W = s ? Wq1 : Wq0;
    WqkvT[idx] = (bf16)W[ci * 1536 + co];
  } else {
    const int j = idx - NQ;
    if (j < 2 * 512 * 512) {
      const int s = j / (512 * 512), r = j % (512 * 512);
      const int co = r / 512, ci = r % 512;
      const float* W = s ? Wp1 : Wp0;
      WpT[j] = (bf16)W[ci * 512 + co];
    }
  }
}

// ---------------------------------------------------------------------------
// Kernel 1: QKV GEMM.  Y = x(4096x512,f32) @ Wqkv(512x1536), bf16 WMMA.
// 128x128 tile, 256 threads = 8 waves (4x2), LDS-staged, f32->bf16 on load.
// Epilogue scatters q,k channel-major and v token-major (all bf16).
// ---------------------------------------------------------------------------
__global__ void __launch_bounds__(256) k_qkv(const float* __restrict__ x0,
                                             const float* __restrict__ x1,
                                             const bf16* __restrict__ WqkvT,
                                             bf16* __restrict__ qT,
                                             bf16* __restrict__ kT,
                                             bf16* __restrict__ vtk) {
  __shared__ alignas(32) bf16 As[128 * 32];
  __shared__ alignas(32) bf16 Bs[128 * 32];
  const int z = blockIdx.z, s = z >> 3, b = z & 7;
  const float* x = (s ? x1 : x0) + (size_t)b * 4096 * 512;
  const bf16* Wt = WqkvT + (size_t)s * 1536 * 512;
  const int m0 = blockIdx.x * 128;  // token tile
  const int c0 = blockIdx.y * 128;  // qkv-channel tile in [0,1536)
  const int tid = threadIdx.x, lane = tid & 31, w = tid >> 5;
  const int wr = w & 3, wc = w >> 2;            // 4x2 wave grid, 32x64 each
  const int arow = tid >> 1, acol = (tid & 1) * 16;

  v8f acc[2][4];
  #pragma unroll
  for (int i = 0; i < 2; ++i)
    #pragma unroll
    for (int j = 0; j < 4; ++j) acc[i][j] = zero8();

  for (int k0 = 0; k0 < 512; k0 += 32) {
    {  // stage A (f32 -> bf16)
      const float* src = x + (size_t)(m0 + arow) * 512 + k0 + acol;
      v4f f0 = *(const v4f*)(src);
      v4f f1 = *(const v4f*)(src + 4);
      v4f f2 = *(const v4f*)(src + 8);
      v4f f3 = *(const v4f*)(src + 12);
      v8bf h0, h1;
      #pragma unroll
      for (int i = 0; i < 4; ++i) {
        h0[i] = (bf16)f0[i]; h0[i + 4] = (bf16)f1[i];
        h1[i] = (bf16)f2[i]; h1[i + 4] = (bf16)f3[i];
      }
      bf16* dst = &As[arow * 32 + acol];
      *(v8bf*)dst = h0;
      *(v8bf*)(dst + 8) = h1;
    }
    {  // stage B (already bf16, transposed)
      const bf16* src = Wt + (size_t)(c0 + arow) * 512 + k0 + acol;
      bf16* dst = &Bs[arow * 32 + acol];
      *(v8bf*)dst = *(const v8bf*)src;
      *(v8bf*)(dst + 8) = *(const v8bf*)(src + 8);
    }
    __syncthreads();
    v16bf af[2], bfb[4];
    #pragma unroll
    for (int i = 0; i < 2; ++i) af[i] = frag_a(&As[(wr * 32 + i * 16) * 32], 32, lane);
    #pragma unroll
    for (int j = 0; j < 4; ++j) bfb[j] = frag_b(&Bs[(wc * 64 + j * 16) * 32], 32, lane);
    #pragma unroll
    for (int i = 0; i < 2; ++i)
      #pragma unroll
      for (int j = 0; j < 4; ++j) acc[i][j] = wmma_bf16(af[i], bfb[j], acc[i][j]);
    __syncthreads();
  }

  bf16* q = qT + (size_t)(s * 8 + b) * 512 * 4096;
  bf16* kk = kT + (size_t)(s * 8 + b) * 512 * 4096;
  bf16* vt = vtk + (size_t)(s * 8 + b) * 4096 * 512;
  #pragma unroll
  for (int i = 0; i < 2; ++i) {
    const int tb = m0 + wr * 32 + i * 16 + 8 * (lane >> 4);  // 8 consecutive tokens
    #pragma unroll
    for (int j = 0; j < 4; ++j) {
      const int c = c0 + wc * 64 + j * 16 + (lane & 15);
      v8bf hv;
      #pragma unroll
      for (int r = 0; r < 8; ++r) hv[r] = (bf16)acc[i][j][r];
      if (c < 512) {                        // q: channel-major, b128 store
        *(v8bf*)(q + (size_t)c * 4096 + tb) = hv;
      } else if (c < 1024) {                // k: channel-major
        *(v8bf*)(kk + (size_t)(c - 512) * 4096 + tb) = hv;
      } else {                              // v: token-major (32B-coalesced per half-wave)
        const int cc = c - 1024;
        #pragma unroll
        for (int r = 0; r < 8; ++r) vt[(size_t)(tb + r) * 512 + cc] = hv[r];
      }
    }
  }
}

// ---------------------------------------------------------------------------
// Kernel 2: inverse L2 norms over the token dim (4096) for q and k rows.
// grid (512 rows, 8 batch, 4 = s*2+which), 256 threads per row.
// ---------------------------------------------------------------------------
__global__ void __launch_bounds__(256) k_norm(const bf16* __restrict__ qT,
                                              const bf16* __restrict__ kT,
                                              float* __restrict__ invq,
                                              float* __restrict__ invk) {
  const int z = blockIdx.z, s = z >> 1, which = z & 1;
  const int b = blockIdx.y, row = blockIdx.x;  // row = h*64+dd in [0,512)
  const bf16* base = (which ? kT : qT) + ((size_t)(s * 8 + b) * 512 + row) * 4096;
  const int tid = threadIdx.x;
  const bf16* p = base + tid * 16;
  v8bf a = *(const v8bf*)p, c = *(const v8bf*)(p + 8);
  float sum = 0.f;
  #pragma unroll
  for (int i = 0; i < 8; ++i) {
    float f = (float)a[i]; sum += f * f;
    f = (float)c[i];       sum += f * f;
  }
  __shared__ float red[256];
  red[tid] = sum;
  __syncthreads();
  for (int off = 128; off > 0; off >>= 1) {
    if (tid < off) red[tid] += red[tid + off];
    __syncthreads();
  }
  if (tid == 0) {
    const float n = fmaxf(sqrtf(red[0]), 1e-12f);
    float* out = which ? invk : invq;
    out[(size_t)(s * 8 + b) * 512 + row] = 1.0f / n;
  }
}

// ---------------------------------------------------------------------------
// Kernel 3: logits S = q_s @ k_{1-s}^T over N=4096, scale by invq*invk*temp,
// softmax over e, store attn bf16 [s][b][h][64][64].
// 1 block per (s,b,h): 512 threads = 16 waves, one 16x16 tile each, K-loop 128.
// ---------------------------------------------------------------------------
__global__ void __launch_bounds__(512) k_attn(const bf16* __restrict__ qT,
                                              const bf16* __restrict__ kT,
                                              const float* __restrict__ invq,
                                              const float* __restrict__ invk,
                                              const float* __restrict__ temp0,
                                              const float* __restrict__ temp1,
                                              bf16* __restrict__ attn) {
  __shared__ alignas(32) float S_l[64 * 64];
  const int blk = blockIdx.x;
  const int h = blk & 7, b = (blk >> 3) & 7, s = blk >> 6;
  const int so = s, ko = 1 - s;  // q from stream s, k from opposite stream
  const bf16* q  = qT + (((size_t)(so * 8 + b) * 512) + h * 64) * 4096;
  const bf16* kp = kT + (((size_t)(ko * 8 + b) * 512) + h * 64) * 4096;
  const int tid = threadIdx.x, lane = tid & 31, w = tid >> 5;
  const int mi = w >> 2, nj = w & 3;

  v8f acc = zero8();
  for (int k0 = 0; k0 < 4096; k0 += 32) {
    v16bf a = frag_a(q + (size_t)(mi * 16) * 4096 + k0, 4096, lane);
    v16bf bb = frag_b(kp + (size_t)(nj * 16) * 4096 + k0, 4096, lane);
    acc = wmma_bf16(a, bb, acc);
  }

  const float* iq = invq + (size_t)(so * 8 + b) * 512 + h * 64;
  const float* ik = invk + (size_t)(ko * 8 + b) * 512 + h * 64;
  const float t = (s == 0 ? temp0 : temp1)[h];
  const int n = nj * 16 + (lane & 15);
  const float ikn = ik[n] * t;
  #pragma unroll
  for (int r = 0; r < 8; ++r) {
    const int m = mi * 16 + r + 8 * (lane >> 4);
    S_l[m * 64 + n] = acc[r] * iq[m] * ikn;
  }
  __syncthreads();

  if (tid < 64) {  // softmax over row tid
    float mx = -3.4e38f;
    for (int e = 0; e < 64; ++e) mx = fmaxf(mx, S_l[tid * 64 + e]);
    float sum = 0.f;
    for (int e = 0; e < 64; ++e) {
      const float v = __expf(S_l[tid * 64 + e] - mx);
      S_l[tid * 64 + e] = v;
      sum += v;
    }
    const float inv = 1.0f / sum;
    bf16* op = attn + (size_t)blk * 4096;
    for (int e = 0; e < 64; ++e) op[tid * 64 + e] = (bf16)(S_l[tid * 64 + e] * inv);
  }
}

// ---------------------------------------------------------------------------
// Kernel 4: out2[n][h*64+dd] = sum_e v_{1-s}[n][h*64+e] * attn_s[dd][e].
// A = v token-major (lda 512), Bt = attn row-major (ldb 64).  K=64 -> 2 steps.
// grid (32 token tiles, 8 heads, 16 = s*8+b); 256 threads = 8 waves (m-tiles).
// ---------------------------------------------------------------------------
__global__ void __launch_bounds__(256) k_attnout(const bf16* __restrict__ vtk,
                                                 const bf16* __restrict__ attn,
                                                 bf16* __restrict__ out2) {
  const int z = blockIdx.z, s = z >> 3, b = z & 7;
  const int h = blockIdx.y, m0 = blockIdx.x * 128;
  const bf16* A  = vtk + (size_t)((1 - s) * 8 + b) * 4096 * 512 + h * 64;
  const bf16* Bt = attn + (size_t)((s * 8 + b) * 8 + h) * 4096;
  bf16* O = out2 + (size_t)(s * 8 + b) * 4096 * 512 + h * 64;
  const int tid = threadIdx.x, lane = tid & 31, w = tid >> 5;

  v8f acc[4];
  #pragma unroll
  for (int j = 0; j < 4; ++j) acc[j] = zero8();
  #pragma unroll
  for (int k0 = 0; k0 < 64; k0 += 32) {
    v16bf a = frag_a(A + (size_t)(m0 + w * 16) * 512 + k0, 512, lane);
    #pragma unroll
    for (int j = 0; j < 4; ++j) {
      v16bf bb = frag_b(Bt + (j * 16) * 64 + k0, 64, lane);
      acc[j] = wmma_bf16(a, bb, acc[j]);
    }
  }
  const int tb = m0 + w * 16 + 8 * (lane >> 4);
  #pragma unroll
  for (int j = 0; j < 4; ++j) {
    const int c = j * 16 + (lane & 15);
    #pragma unroll
    for (int r = 0; r < 8; ++r) O[(size_t)(tb + r) * 512 + c] = (bf16)acc[j][r];
  }
}

// ---------------------------------------------------------------------------
// Kernel 5: projection.  out = out2(bf16) @ Wp + bias, f32 result to d_out.
// Same tiling as k_qkv (128x128, LDS-staged).
// ---------------------------------------------------------------------------
__global__ void __launch_bounds__(256) k_proj(const bf16* __restrict__ out2,
                                              const bf16* __restrict__ WpT,
                                              const float* __restrict__ bias0,
                                              const float* __restrict__ bias1,
                                              float* __restrict__ out) {
  __shared__ alignas(32) bf16 As[128 * 32];
  __shared__ alignas(32) bf16 Bs[128 * 32];
  const int z = blockIdx.z, s = z >> 3, b = z & 7;
  const bf16* A = out2 + (size_t)(s * 8 + b) * 4096 * 512;
  const bf16* Wt = WpT + (size_t)s * 512 * 512;
  const float* bias = s ? bias1 : bias0;
  float* O = out + (size_t)(s * 8 + b) * 4096 * 512;
  const int m0 = blockIdx.x * 128, c0 = blockIdx.y * 128;
  const int tid = threadIdx.x, lane = tid & 31, w = tid >> 5;
  const int wr = w & 3, wc = w >> 2;
  const int arow = tid >> 1, acol = (tid & 1) * 16;

  v8f acc[2][4];
  #pragma unroll
  for (int i = 0; i < 2; ++i)
    #pragma unroll
    for (int j = 0; j < 4; ++j) acc[i][j] = zero8();

  for (int k0 = 0; k0 < 512; k0 += 32) {
    {
      const bf16* src = A + (size_t)(m0 + arow) * 512 + k0 + acol;
      bf16* dst = &As[arow * 32 + acol];
      *(v8bf*)dst = *(const v8bf*)src;
      *(v8bf*)(dst + 8) = *(const v8bf*)(src + 8);
    }
    {
      const bf16* src = Wt + (size_t)(c0 + arow) * 512 + k0 + acol;
      bf16* dst = &Bs[arow * 32 + acol];
      *(v8bf*)dst = *(const v8bf*)src;
      *(v8bf*)(dst + 8) = *(const v8bf*)(src + 8);
    }
    __syncthreads();
    v16bf af[2], bfb[4];
    #pragma unroll
    for (int i = 0; i < 2; ++i) af[i] = frag_a(&As[(wr * 32 + i * 16) * 32], 32, lane);
    #pragma unroll
    for (int j = 0; j < 4; ++j) bfb[j] = frag_b(&Bs[(wc * 64 + j * 16) * 32], 32, lane);
    #pragma unroll
    for (int i = 0; i < 2; ++i)
      #pragma unroll
      for (int j = 0; j < 4; ++j) acc[i][j] = wmma_bf16(af[i], bfb[j], acc[i][j]);
    __syncthreads();
  }

  #pragma unroll
  for (int i = 0; i < 2; ++i) {
    const int tb = m0 + wr * 32 + i * 16 + 8 * (lane >> 4);
    #pragma unroll
    for (int j = 0; j < 4; ++j) {
      const int c = c0 + wc * 64 + j * 16 + (lane & 15);
      const float bi = bias[c];
      #pragma unroll
      for (int r = 0; r < 8; ++r) O[(size_t)(tb + r) * 512 + c] = acc[i][j][r] + bi;
    }
  }
}

// ---------------------------------------------------------------------------
extern "C" void kernel_launch(void* const* d_in, const int* in_sizes, int n_in,
                              void* d_out, int out_size, void* d_ws, size_t ws_size,
                              hipStream_t stream) {
  const float* x0  = (const float*)d_in[0];  // x        (8,4096,512)
  const float* x1  = (const float*)d_in[1];  // x_d
  const float* Wq0 = (const float*)d_in[2];  // W_qkv_rgb (512,1536)
  const float* Wq1 = (const float*)d_in[3];  // W_qkv_T
  const float* t0  = (const float*)d_in[4];  // temperature_rgb (8)
  const float* t1  = (const float*)d_in[5];  // temperature_T
  const float* Wp0 = (const float*)d_in[6];  // W_proj_rgb (512,512)
  const float* bp0 = (const float*)d_in[7];  // b_proj_rgb (512)
  const float* Wp1 = (const float*)d_in[8];  // W_proj_T
  const float* bp1 = (const float*)d_in[9];  // b_proj_T
  float* out = (float*)d_out;                // [out_rgb | out_T], each (8,4096,512) f32

  char* ws = (char*)d_ws;
  bf16* qT    = (bf16*)(ws);
  bf16* kT    = (bf16*)(ws + ((size_t)64 << 20));
  bf16* vtk   = (bf16*)(ws + ((size_t)128 << 20));
  bf16* out2  = qT;  // qT is dead after k_attn; reuse its 64MB for out2
  bf16* WqkvT = (bf16*)(ws + ((size_t)192 << 20));
  bf16* WpT   = (bf16*)(ws + ((size_t)196 << 20));
  bf16* attn  = (bf16*)(ws + ((size_t)198 << 20));
  float* invq = (float*)(ws + ((size_t)200 << 20));
  float* invk = (float*)(ws + ((size_t)200 << 20) + (64 << 10));

  k_prep<<<8192, 256, 0, stream>>>(Wq0, Wq1, Wp0, Wp1, WqkvT, WpT);
  k_qkv<<<dim3(32, 12, 16), 256, 0, stream>>>(x0, x1, WqkvT, qT, kT, vtk);
  k_norm<<<dim3(512, 8, 4), 256, 0, stream>>>(qT, kT, invq, invk);
  k_attn<<<128, 512, 0, stream>>>(qT, kT, invq, invk, t0, t1, attn);
  k_attnout<<<dim3(32, 8, 16), 256, 0, stream>>>(vtk, attn, out2);
  k_proj<<<dim3(32, 4, 16), 256, 0, stream>>>(out2, WpT, bp0, bp1, out);
}
